// GATModel_67284957659783
// MI455X (gfx1250) — compile-verified, matrix operand order checked by reading, not run on hardware
//
#include <hip/hip_runtime.h>
#include <hip/hip_bf16.h>

// ---------- types ----------
typedef __attribute__((ext_vector_type(16))) __bf16 bf16x16;
typedef __attribute__((ext_vector_type(8)))  float  f32x8;
typedef __attribute__((ext_vector_type(4)))  unsigned int u32x4;

union Frag16 { bf16x16 v; u32x4 u[2]; };

#define HID 64

// ---------- small helpers ----------
__device__ __forceinline__ void atomicMaxFloat(float* addr, float val) {
    if (val >= 0.0f) atomicMax((int*)addr, __float_as_int(val));
    else             atomicMin((unsigned int*)addr, __float_as_uint(val));
}

__device__ __forceinline__ float leaky02(float v) {
    return v >= 0.0f ? v : 0.2f * v;
}

// ---------- generic fill ----------
__global__ void fill_f32_k(float* __restrict__ p, float v, long long n) {
    long long t = (long long)blockIdx.x * blockDim.x + threadIdx.x;
    if (t < n) p[t] = v;
}

// ---------- f32 -> bf16 convert ----------
__global__ void cvt_bf16_k(const float* __restrict__ x, __bf16* __restrict__ xb, long long n) {
    long long t = (long long)blockIdx.x * blockDim.x + threadIdx.x;
    if (t < n) xb[t] = (__bf16)x[t];
}

// ---------- pack W[K,Nout] (f32, row-major) into WMMA B-fragment layout (bf16) ----------
// Fragment (kt, nt): per lane l, col n = nt*16 + (l&15), kh = (l>>4)*8.
// dst[j]   = W[kt*32 + kh + j      ][n]   j=0..7
// dst[j+8] = W[kt*32 + 16 + kh + j ][n]   j=0..7
__global__ void pack_w_k(const float* __restrict__ W, __bf16* __restrict__ Wp, int K, int Nout) {
    int t = blockIdx.x * blockDim.x + threadIdx.x;
    int NT = Nout >> 4, KT = K >> 5;
    int total = KT * NT * 32;
    if (t >= total) return;
    int lane = t & 31;
    int frag = t >> 5;
    int nt = frag % NT, kt = frag / NT;
    int n  = nt * 16 + (lane & 15);
    int kh = (lane >> 4) * 8;
    __bf16* dst = Wp + (size_t)t * 16;
    for (int j = 0; j < 8; ++j) {
        dst[j]     = (__bf16)W[(size_t)(kt * 32 + kh + j) * Nout + n];
        dst[j + 8] = (__bf16)W[(size_t)(kt * 32 + 16 + kh + j) * Nout + n];
    }
}

// ---------- bf16 GEMM via v_wmma_f32_16x16x32_bf16: C[M,64] = A[M,K] x Bpacked ----------
// One wave owns a full 16-row stripe (all 4 n-tiles of the 64-wide output):
//  - A fragment loaded once per k-step, reused by 4 independent WMMAs
//  - K is a template constant -> loop fully unrolls, loads clause together
//  - tile id is wave-uniform so EXEC stays all-ones for WMMA
template<int K>
__global__ void __launch_bounds__(128)
gemm_wmma_bf16_k(const __bf16* __restrict__ A,
                 const __bf16* __restrict__ Bp,
                 float* __restrict__ C,
                 int M) {
    constexpr int KT = K / 32;
    constexpr int NT = 4;                   // Nout = 64
    int lane = threadIdx.x & 31;
    int wave = threadIdx.x >> 5;
    int MT = (M + 15) >> 4;
    int mt = blockIdx.x * (blockDim.x >> 5) + wave;
    if (mt >= MT) return;                   // whole-wave uniform exit

    int row = mt * 16 + (lane & 15);
    if (row >= M) row = M - 1;              // clamp (garbage rows never stored)
    int kh = (lane >> 4) * 8;

    f32x8 c[NT];
    #pragma unroll
    for (int nt = 0; nt < NT; ++nt)
        c[nt] = f32x8{0.f, 0.f, 0.f, 0.f, 0.f, 0.f, 0.f, 0.f};

    const __bf16* arow = A + (size_t)row * K + kh;
    #pragma unroll
    for (int kt = 0; kt < KT; ++kt) {
        Frag16 a;
        a.u[0] = *(const u32x4*)(arow + kt * 32);
        a.u[1] = *(const u32x4*)(arow + kt * 32 + 16);
        #pragma unroll
        for (int nt = 0; nt < NT; ++nt) {
            Frag16 b;
            const __bf16* bp = Bp + ((size_t)(kt * NT + nt) * 32 + lane) * 16;
            b.u[0] = *(const u32x4*)(bp);
            b.u[1] = *(const u32x4*)(bp + 8);
            c[nt] = __builtin_amdgcn_wmma_f32_16x16x32_bf16(
                        /*neg_a=*/false, a.v, /*neg_b=*/false, b.v,
                        /*c_mod=*/(short)0, c[nt], /*reuse_a=*/false, /*reuse_b=*/false);
        }
    }

    // C layout: lane -> col = nt*16 + (l&15); VGPR r -> row mt*16 + r + 8*(l>>4)
    int mrow = mt * 16 + 8 * (lane >> 4);
    #pragma unroll
    for (int nt = 0; nt < NT; ++nt) {
        int col = nt * 16 + (lane & 15);
        #pragma unroll
        for (int r = 0; r < 8; ++r) {
            int rr = mrow + r;
            if (rr < M) C[(size_t)rr * HID + col] = c[nt][r];
        }
    }
}

// ---------- attention logit halves: al_src[i] = h[i]·a_src, al_dst[i] = h[i]·a_dst ----------
__global__ void al_k(const float* __restrict__ h,
                     const float* __restrict__ a_s, const float* __restrict__ a_d,
                     float* __restrict__ alS, float* __restrict__ alD, long long N) {
    long long i = (long long)blockIdx.x * blockDim.x + threadIdx.x;
    if (i >= N) return;
    const float* row = h + (size_t)i * HID;
    float s = 0.f, d = 0.f;
    for (int f = 0; f < HID; ++f) { float v = row[f]; s += v * a_s[f]; d += v * a_d[f]; }
    alS[i] = s; alD[i] = d;
}

// ---------- edge passes (edge list = [E real edges] ++ [N self-loops]) ----------
__device__ __forceinline__ void edge_nodes(const long long* __restrict__ src,
                                           const long long* __restrict__ dst,
                                           long long E, long long e, int& s, int& d) {
    if (e < E) { s = (int)src[e]; d = (int)dst[e]; }
    else       { s = d = (int)(e - E); }
}

__global__ void edge_max_k(const long long* __restrict__ src, const long long* __restrict__ dst,
                           long long E, long long ET,
                           const float* __restrict__ alS, const float* __restrict__ alD,
                           float* __restrict__ m) {
    long long e = (long long)blockIdx.x * blockDim.x + threadIdx.x;
    if (e >= ET) return;
    int s, d; edge_nodes(src, dst, E, e, s, d);
    atomicMaxFloat(&m[d], leaky02(alS[s] + alD[d]));
}

__global__ void edge_expsum_k(const long long* __restrict__ src, const long long* __restrict__ dst,
                              long long E, long long ET,
                              const float* __restrict__ alS, const float* __restrict__ alD,
                              const float* __restrict__ m, float* __restrict__ ssum) {
    long long e = (long long)blockIdx.x * blockDim.x + threadIdx.x;
    if (e >= ET) return;
    int s, d; edge_nodes(src, dst, E, e, s, d);
    float v = leaky02(alS[s] + alD[d]);
    atomicAdd(&ssum[d], __expf(v - m[d]));
}

__global__ void edge_coef_k(const long long* __restrict__ src, const long long* __restrict__ dst,
                            long long E, long long ET,
                            const float* __restrict__ alS, const float* __restrict__ alD,
                            const float* __restrict__ m, const float* __restrict__ ssum,
                            float* __restrict__ coef) {
    long long e = (long long)blockIdx.x * blockDim.x + threadIdx.x;
    if (e >= ET) return;
    int s, d; edge_nodes(src, dst, E, e, s, d);
    float v = leaky02(alS[s] + alD[d]);
    coef[e] = __expf(v - m[d]) / ssum[d];
}

// One thread per (edge, feature): 64 consecutive lanes share an edge ->
// coalesced gather of h[src] and coalesced f32 atomics into hout[dst].
__global__ void edge_scatter_k(const long long* __restrict__ src, const long long* __restrict__ dst,
                               long long E, long long ET,
                               const float* __restrict__ coef,
                               const float* __restrict__ h, float* __restrict__ ho) {
    long long t = (long long)blockIdx.x * blockDim.x + threadIdx.x;
    if (t >= ET * HID) return;
    long long e = t >> 6;
    int f = (int)(t & 63);
    int s, d; edge_nodes(src, dst, E, e, s, d);
    atomicAdd(&ho[(size_t)d * HID + f], h[(size_t)s * HID + f] * coef[e]);
}

// ---------- bias + ELU (alpha=1), optional bf16 shadow copy for next-layer WMMA ----------
__global__ void bias_elu_k(float* __restrict__ h, const float* __restrict__ b,
                           __bf16* __restrict__ hb, long long N) {
    long long t = (long long)blockIdx.x * blockDim.x + threadIdx.x;
    if (t >= N * HID) return;
    int f = (int)(t & 63);
    float v = h[t] + b[f];
    v = v > 0.0f ? v : (__expf(v) - 1.0f);
    h[t] = v;
    if (hb) hb[t] = (__bf16)v;
}

// ---------- global mean pool ----------
__global__ void pool_cnt_k(const long long* __restrict__ batch, float* __restrict__ cnt, long long N) {
    long long i = (long long)blockIdx.x * blockDim.x + threadIdx.x;
    if (i >= N) return;
    atomicAdd(&cnt[(int)batch[i]], 1.0f);
}

__global__ void pool_sum_k(const float* __restrict__ h, const long long* __restrict__ batch,
                           float* __restrict__ pooled, long long N) {
    long long t = (long long)blockIdx.x * blockDim.x + threadIdx.x;
    if (t >= N * HID) return;
    long long i = t >> 6;
    int f = (int)(t & 63);
    atomicAdd(&pooled[(size_t)batch[i] * HID + f], h[t]);
}

__global__ void final_k(const float* __restrict__ pooled, const float* __restrict__ cnt,
                        const float* __restrict__ Wfc, const float* __restrict__ bfc,
                        float* __restrict__ out, int G, int OD) {
    int g = blockIdx.x * blockDim.x + threadIdx.x;
    if (g >= G) return;
    float c = cnt[g]; if (c < 1.0f) c = 1.0f;
    for (int o = 0; o < OD; ++o) {
        float acc = 0.f;
        for (int f = 0; f < HID; ++f)
            acc += (pooled[(size_t)g * HID + f] / c) * Wfc[(size_t)f * OD + o];
        out[(size_t)g * OD + o] = acc + bfc[o];
    }
}

// ---------- host launcher ----------
extern "C" void kernel_launch(void* const* d_in, const int* in_sizes, int n_in,
                              void* d_out, int out_size, void* d_ws, size_t ws_size,
                              hipStream_t stream) {
    const float*      x      = (const float*)d_in[0];
    const long long*  ei     = (const long long*)d_in[1];   // [2,E] int64
    const long long*  batch  = (const long long*)d_in[2];
    const float*      W1     = (const float*)d_in[3];
    const float*      a_src1 = (const float*)d_in[4];
    const float*      a_dst1 = (const float*)d_in[5];
    const float*      b1     = (const float*)d_in[6];
    const float*      W2     = (const float*)d_in[7];
    const float*      a_src2 = (const float*)d_in[8];
    const float*      a_dst2 = (const float*)d_in[9];
    const float*      b2     = (const float*)d_in[10];
    const float*      Wfc    = (const float*)d_in[11];
    const float*      bfc    = (const float*)d_in[12];
    float*            out    = (float*)d_out;

    const long long N  = (long long)in_sizes[0] / 128;
    const long long E  = (long long)in_sizes[1] / 2;
    const long long ET = E + N;                      // + self loops
    const int IN = 128, G = 256;
    const int OD = in_sizes[11] / HID;
    const long long* esrc = ei;
    const long long* edst = ei + E;

    // workspace carve-up (256B aligned slices)
    char* ws = (char*)d_ws;
    size_t off = 0;
    auto take = [&](size_t bytes) -> char* {
        char* p = ws + off;
        off = (off + bytes + 255) & ~(size_t)255;
        return p;
    };
    __bf16* xb     = (__bf16*)take((size_t)N * IN * 2);
    __bf16* hactb  = (__bf16*)take((size_t)N * HID * 2);
    float*  hA     = (float*)take((size_t)N * HID * 4);   // GEMM out (pre-attention h)
    float*  hB     = (float*)take((size_t)N * HID * 4);   // attention accumulator / activations
    float*  alS    = (float*)take((size_t)N * 4);
    float*  alD    = (float*)take((size_t)N * 4);
    float*  mbuf   = (float*)take((size_t)N * 4);
    float*  sbuf   = (float*)take((size_t)N * 4);
    float*  coef   = (float*)take((size_t)ET * 4);
    __bf16* Wp1    = (__bf16*)take((size_t)(IN / 32) * (HID / 16) * 32 * 16 * 2);
    __bf16* Wp2    = (__bf16*)take((size_t)(HID / 32) * (HID / 16) * 32 * 16 * 2);
    float*  pooled = (float*)take((size_t)G * HID * 4);
    float*  cnt    = (float*)take((size_t)G * 4);
    (void)ws_size; (void)n_in; (void)out_size;

    const int B = 256;
    auto nb = [](long long n, int b) -> unsigned { return (unsigned)((n + b - 1) / b); };

    const float NEG_INF = -__builtin_inff();
    const long long MT = (N + 15) / 16;
    const unsigned gemmBlocks = nb(MT, 4);         // 4 waves (16-row stripes) per 128-thread block

    // ---- prep: bf16 copy of x, pack weights into B-fragment layout ----
    cvt_bf16_k<<<nb(N * IN, B), B, 0, stream>>>(x, xb, N * IN);
    pack_w_k<<<nb((IN / 32) * (HID / 16) * 32, B), B, 0, stream>>>(W1, Wp1, IN, HID);
    pack_w_k<<<nb((HID / 32) * (HID / 16) * 32, B), B, 0, stream>>>(W2, Wp2, HID, HID);

    // ---- layer 1 ----
    gemm_wmma_bf16_k<128><<<gemmBlocks, 128, 0, stream>>>(xb, Wp1, hA, (int)N);
    al_k<<<nb(N, B), B, 0, stream>>>(hA, a_src1, a_dst1, alS, alD, N);
    fill_f32_k<<<nb(N, B), B, 0, stream>>>(mbuf, NEG_INF, N);
    fill_f32_k<<<nb(N, B), B, 0, stream>>>(sbuf, 0.f, N);
    fill_f32_k<<<nb(N * HID, B), B, 0, stream>>>(hB, 0.f, N * HID);
    edge_max_k   <<<nb(ET, B), B, 0, stream>>>(esrc, edst, E, ET, alS, alD, mbuf);
    edge_expsum_k<<<nb(ET, B), B, 0, stream>>>(esrc, edst, E, ET, alS, alD, mbuf, sbuf);
    edge_coef_k  <<<nb(ET, B), B, 0, stream>>>(esrc, edst, E, ET, alS, alD, mbuf, sbuf, coef);
    edge_scatter_k<<<nb(ET * HID, B), B, 0, stream>>>(esrc, edst, E, ET, coef, hA, hB);
    bias_elu_k<<<nb(N * HID, B), B, 0, stream>>>(hB, b1, hactb, N);

    // ---- layer 2 ----
    gemm_wmma_bf16_k<64><<<gemmBlocks, 128, 0, stream>>>(hactb, Wp2, hA, (int)N);
    al_k<<<nb(N, B), B, 0, stream>>>(hA, a_src2, a_dst2, alS, alD, N);
    fill_f32_k<<<nb(N, B), B, 0, stream>>>(mbuf, NEG_INF, N);
    fill_f32_k<<<nb(N, B), B, 0, stream>>>(sbuf, 0.f, N);
    fill_f32_k<<<nb(N * HID, B), B, 0, stream>>>(hB, 0.f, N * HID);
    edge_max_k   <<<nb(ET, B), B, 0, stream>>>(esrc, edst, E, ET, alS, alD, mbuf);
    edge_expsum_k<<<nb(ET, B), B, 0, stream>>>(esrc, edst, E, ET, alS, alD, mbuf, sbuf);
    edge_coef_k  <<<nb(ET, B), B, 0, stream>>>(esrc, edst, E, ET, alS, alD, mbuf, sbuf, coef);
    edge_scatter_k<<<nb(ET * HID, B), B, 0, stream>>>(esrc, edst, E, ET, coef, hA, hB);
    bias_elu_k<<<nb(N * HID, B), B, 0, stream>>>(hB, b2, (__bf16*)nullptr, N);

    // ---- global mean pool + head ----
    fill_f32_k<<<nb((long long)G * HID, B), B, 0, stream>>>(pooled, 0.f, (long long)G * HID);
    fill_f32_k<<<nb(G, B), B, 0, stream>>>(cnt, 0.f, G);
    pool_cnt_k<<<nb(N, B), B, 0, stream>>>(batch, cnt, N);
    pool_sum_k<<<nb(N * HID, B), B, 0, stream>>>(hB, batch, pooled, N);
    final_k<<<1, G, 0, stream>>>(pooled, cnt, Wfc, bfc, out, G, OD);
}
